// Attention_12034498363513
// MI455X (gfx1250) — compile-verified, matrix operand order checked by reading, not run on hardware
//
#include <hip/hip_runtime.h>
#include <hip/hip_bf16.h>

// ---------------------------------------------------------------------------
// Fastformer additive-attention layer for MI455X (gfx1250, wave32, WMMA).
// B=8, S=2048, D=1024, H=16, HD=64.
// GEMMs: v_wmma_f32_16x16x32_f16, double-buffered LDS, async global->LDS
// staging (ASYNCcnt), L2 prefetch via global_prefetch_b8.
// ---------------------------------------------------------------------------

typedef _Float16 half_t;
typedef __attribute__((ext_vector_type(16))) _Float16 v16h;
typedef __attribute__((ext_vector_type(8)))  float    v8f;

union FragH { v16h v; unsigned int u[8]; };
union FragF { v8f  v; float f[8]; };

#define TM 128
#define TN 128
#define TK 32
#define LDS_AS 40   // LDS row stride in halves (32 data + 8 pad) -> 80B, 16B aligned
#define LDS_BS 40

// K mapping for CDNA5 16-bit WMMA fragments (ISA 7.12.2):
// VGPR j, lane L:  K = (L>=16 ? 8 : 0) + (j>=4 ? 16 : 0) + 2*(j&3) (+0/+1 packed)
__device__ __forceinline__ int frag_k(int j, int kb) {
    return kb + ((j >> 2) << 4) + ((j & 3) << 1);
}

// --- CDNA5 async global->LDS copy (tracked by ASYNCcnt) --------------------
typedef __attribute__((address_space(3))) void lds_void;

__device__ __forceinline__ unsigned lds_addr_of(const void* p) {
    // generic -> LDS address space; VGPR holds byte offset in wave's LDS
    return (unsigned)(unsigned long long)(lds_void*)p;
}

__device__ __forceinline__ void async_b128(const half_t* gsrc, unsigned lds_off) {
    // GLOBAL_LOAD_ASYNC_TO_LDS_B128: vdst = LDS byte address, vaddr = 64b global
    asm volatile("global_load_async_to_lds_b128 %0, %1, off"
                 :: "v"(lds_off), "v"(gsrc)
                 : "memory");
}

__device__ __forceinline__ void wait_asynccnt0() {
    asm volatile("s_wait_asynccnt 0x0" ::: "memory");
}

// ---------------------------------------------------------------------------
// Generic 128x128 tiled f16 WMMA GEMM:  C = A[M,K] @ Bw[K,N] (+ epilogue)
// mode 0: C = acc + bias(qkv-split b0/b1/b2)          (QKV projection)
// mode 1: C = acc + b0[col] + resid[row,col]          (Wo projection + residual)
// Software-pipelined with last-iteration peel: tile k+1 staged (A async->LDS,
// B via regs) while tile k feeds the 8 WMMAs/wave; tile k+2 prefetched to L2.
// ---------------------------------------------------------------------------
__launch_bounds__(256)
__global__ void wmma_gemm_f16(const half_t* __restrict__ A, int lda,
                              const half_t* __restrict__ Bw, int ldb,
                              float* __restrict__ C, int ldc, int K,
                              const float* __restrict__ b0,
                              const float* __restrict__ b1,
                              const float* __restrict__ b2,
                              const float* __restrict__ resid,
                              int mode)
{
    __shared__ __align__(16) half_t Ah[2][TM * LDS_AS];
    __shared__ __align__(16) half_t Bt[2][TN * LDS_BS];   // transposed: Bt[n][k]

    const int tid  = threadIdx.x;
    const int wave = tid >> 5;
    const int lane = tid & 31;
    const int wm   = wave >> 1;   // 0..3 -> 32 rows each
    const int wn   = wave & 1;    // 0..1 -> 64 cols each
    const int m0   = blockIdx.x * TM;
    const int n0   = blockIdx.y * TN;

    FragF acc[2][4];
    #pragma unroll
    for (int i = 0; i < 2; ++i)
        #pragma unroll
        for (int j = 0; j < 4; ++j)
            #pragma unroll
            for (int e = 0; e < 8; ++e) acc[i][j].f[e] = 0.0f;

    const int lm = lane & 15;
    const int kb = (lane >> 4) * 8;

    // per-thread staging roles
    const int arow = tid >> 1;
    const int ahb  = (tid & 1) * 16;
    const half_t* Abase = A + (size_t)(m0 + arow) * lda + ahb;
    const int bk  = tid & 31;
    const int bnc = (tid >> 5) * 16;
    const half_t* Bbase = Bw + (size_t)bk * ldb + n0 + bnc;

    __align__(16) half_t btmp[16];
    const int KT = K / TK;

    // --- prologue: stage tile 0 into buffer 0 ---
    async_b128(Abase,     lds_addr_of(&Ah[0][arow * LDS_AS + ahb]));
    async_b128(Abase + 8, lds_addr_of(&Ah[0][arow * LDS_AS + ahb + 8]));
    *(uint4*)&btmp[0] = *(const uint4*)(Bbase);
    *(uint4*)&btmp[8] = *(const uint4*)(Bbase + 8);
    #pragma unroll
    for (int j = 0; j < 16; ++j)
        Bt[0][(bnc + j) * LDS_BS + bk] = btmp[j];
    wait_asynccnt0();
    __syncthreads();

    // --- main loop: always stages tile kt+1 (no branches, no dummy init) ---
    for (int kt = 0; kt < KT - 1; ++kt) {
        const int cur = kt & 1;
        const int nxt = cur ^ 1;

        // A tile k+1: async DMA straight into the other LDS buffer
        const half_t* An = Abase + (size_t)(kt + 1) * TK;
        async_b128(An,     lds_addr_of(&Ah[nxt][arow * LDS_AS + ahb]));
        async_b128(An + 8, lds_addr_of(&Ah[nxt][arow * LDS_AS + ahb + 8]));
        // B tile k+1: into registers (transpose happens on LDS store)
        const half_t* Bn = Bbase + (size_t)(kt + 1) * TK * ldb;
        const uint4 bp0 = *(const uint4*)(Bn);
        const uint4 bp1 = *(const uint4*)(Bn + 8);
        if (kt + 2 < KT) {
            __builtin_prefetch(Abase + (size_t)(kt + 2) * TK, 0, 1);
            __builtin_prefetch(Bbase + (size_t)(kt + 2) * TK * ldb, 0, 1);
        }

        // --- fragments from current buffer ---
        FragH afrag[2], bfrag[4];
        #pragma unroll
        for (int i = 0; i < 2; ++i) {
            const half_t* base = &Ah[cur][(wm * 32 + i * 16 + lm) * LDS_AS];
            #pragma unroll
            for (int j = 0; j < 8; ++j)
                afrag[i].u[j] = *(const unsigned int*)(base + frag_k(j, kb));
        }
        #pragma unroll
        for (int j4 = 0; j4 < 4; ++j4) {
            const half_t* base = &Bt[cur][(wn * 64 + j4 * 16 + lm) * LDS_BS];
            #pragma unroll
            for (int j = 0; j < 8; ++j)
                bfrag[j4].u[j] = *(const unsigned int*)(base + frag_k(j, kb));
        }

        // drain B regs into next buffer while WMMAs below execute
        *(uint4*)&btmp[0] = bp0;
        *(uint4*)&btmp[8] = bp1;
        #pragma unroll
        for (int j = 0; j < 16; ++j)
            Bt[nxt][(bnc + j) * LDS_BS + bk] = btmp[j];

        #pragma unroll
        for (int i = 0; i < 2; ++i)
            #pragma unroll
            for (int j = 0; j < 4; ++j)
                acc[i][j].v = __builtin_amdgcn_wmma_f32_16x16x32_f16(
                    false, afrag[i].v, false, bfrag[j].v,
                    (short)0, acc[i][j].v, false, false);

        wait_asynccnt0();
        __syncthreads();
    }

    // --- peeled last K-step: consume only ---
    {
        const int cur = (KT - 1) & 1;
        FragH afrag[2], bfrag[4];
        #pragma unroll
        for (int i = 0; i < 2; ++i) {
            const half_t* base = &Ah[cur][(wm * 32 + i * 16 + lm) * LDS_AS];
            #pragma unroll
            for (int j = 0; j < 8; ++j)
                afrag[i].u[j] = *(const unsigned int*)(base + frag_k(j, kb));
        }
        #pragma unroll
        for (int j4 = 0; j4 < 4; ++j4) {
            const half_t* base = &Bt[cur][(wn * 64 + j4 * 16 + lm) * LDS_BS];
            #pragma unroll
            for (int j = 0; j < 8; ++j)
                bfrag[j4].u[j] = *(const unsigned int*)(base + frag_k(j, kb));
        }
        #pragma unroll
        for (int i = 0; i < 2; ++i)
            #pragma unroll
            for (int j = 0; j < 4; ++j)
                acc[i][j].v = __builtin_amdgcn_wmma_f32_16x16x32_f16(
                    false, afrag[i].v, false, bfrag[j].v,
                    (short)0, acc[i][j].v, false, false);
    }

    // --- epilogue: C/D layout (ISA 7.12.2): VGPR r, lanes<16 M=r, lanes>=16 M=8+r
    #pragma unroll
    for (int i = 0; i < 2; ++i) {
        const int rbase = m0 + wm * 32 + i * 16 + (lane >> 4) * 8;
        #pragma unroll
        for (int j = 0; j < 4; ++j) {
            const int col = n0 + wn * 64 + j * 16 + lm;
            #pragma unroll
            for (int r = 0; r < 8; ++r) {
                const int row = rbase + r;
                float val = acc[i][j].f[r];
                if (mode == 0) {
                    float bias = (col < 1024) ? b0[col]
                               : (col < 2048) ? b1[col - 1024]
                                              : b2[col - 2048];
                    C[(size_t)row * ldc + col] = val + bias;
                } else {
                    C[(size_t)row * ldc + col] =
                        val + b0[col] + resid[(size_t)row * ldc + col];
                }
            }
        }
    }
}

// ---------------------------------------------------------------------------
// Per-head small GEMM: r = u[.,h,:,:] @ Wu + bu ; newr = f16(r + q)
// Block: 128 rows x 64 cols, 8 waves (16 rows each), K=64 (two 32-chunks).
// A tile staged with async global->LDS DMA.
// ---------------------------------------------------------------------------
__launch_bounds__(256)
__global__ void wmma_gemm_wu(const half_t* __restrict__ U,    // [16384,1024] f16
                             const half_t* __restrict__ Wu,   // [64,64] f16
                             const float* __restrict__ bu,    // [64]
                             const float* __restrict__ qkv,   // [16384,3072] f32
                             half_t* __restrict__ NewR)       // [16384,1024] f16
{
    __shared__ __align__(16) half_t Ah[128 * 72];
    __shared__ __align__(16) half_t Bt[64 * 72];

    const int tid  = threadIdx.x;
    const int wave = tid >> 5;
    const int lane = tid & 31;
    const int h    = blockIdx.y;
    const int m0   = blockIdx.x * 128;

    // stage A tile (async DMA): rows m0..m0+127, head slice [h*64, h*64+64)
    {
        const int row = tid >> 1;
        const int hb  = (tid & 1) * 32;
        const half_t* src = U + (size_t)(m0 + row) * 1024 + h * 64 + hb;
        const unsigned l0 = lds_addr_of(&Ah[row * 72 + hb]);
        async_b128(src,      l0);
        async_b128(src + 8,  l0 + 16);
        async_b128(src + 16, l0 + 32);
        async_b128(src + 24, l0 + 48);
    }
    // stage Wu transposed: Bt[n][k]
    {
        const int k  = tid & 63;
        const int nc = (tid >> 6) * 16;
        const half_t* src = Wu + k * 64 + nc;
        __align__(16) half_t tmp[16];
        *(uint4*)&tmp[0] = *(const uint4*)(src);
        *(uint4*)&tmp[8] = *(const uint4*)(src + 8);
        #pragma unroll
        for (int j = 0; j < 16; ++j)
            Bt[(nc + j) * 72 + k] = tmp[j];
    }
    wait_asynccnt0();
    __syncthreads();

    FragF acc[4];
    #pragma unroll
    for (int j = 0; j < 4; ++j)
        #pragma unroll
        for (int e = 0; e < 8; ++e) acc[j].f[e] = 0.0f;

    const int lm = lane & 15;
    const int kb = (lane >> 4) * 8;

    #pragma unroll
    for (int ks = 0; ks < 2; ++ks) {
        FragH af;
        const half_t* abase = &Ah[(wave * 16 + lm) * 72 + ks * 32];
        #pragma unroll
        for (int j = 0; j < 8; ++j)
            af.u[j] = *(const unsigned int*)(abase + frag_k(j, kb));
        #pragma unroll
        for (int j4 = 0; j4 < 4; ++j4) {
            FragH bf;
            const half_t* bbase = &Bt[(j4 * 16 + lm) * 72 + ks * 32];
            #pragma unroll
            for (int j = 0; j < 8; ++j)
                bf.u[j] = *(const unsigned int*)(bbase + frag_k(j, kb));
            acc[j4].v = __builtin_amdgcn_wmma_f32_16x16x32_f16(
                false, af.v, false, bf.v, (short)0, acc[j4].v, false, false);
        }
    }

    const int rbase = m0 + wave * 16 + (lane >> 4) * 8;
    #pragma unroll
    for (int j4 = 0; j4 < 4; ++j4) {
        const int n = j4 * 16 + lm;
        #pragma unroll
        for (int r = 0; r < 8; ++r) {
            const int row = rbase + r;
            float val = acc[j4].f[r] + bu[n] +
                        qkv[(size_t)row * 3072 + h * 64 + n];   // + q
            NewR[(size_t)row * 1024 + h * 64 + n] = (half_t)val;
        }
    }
}

// ---------------------------------------------------------------------------
// Additive-attention pooling: per (b,h) workgroup.
// phase 0: alpha softmax over q -> q_av (in LDS gv)
// phase 1: beta softmax over p=k*q_av -> p_av (to global)
// ---------------------------------------------------------------------------
__launch_bounds__(256)
__global__ void attn_pool(const float* __restrict__ qkv,
                          const float* __restrict__ mask,
                          const float* __restrict__ wa, const float* __restrict__ ba,
                          const float* __restrict__ wb, const float* __restrict__ bb,
                          float* __restrict__ pav)
{
    __shared__ float sc[2048];
    __shared__ float red[256];
    __shared__ float wv[64];
    __shared__ float gv[64];

    const int tid = threadIdx.x;
    const int b   = blockIdx.x >> 4;
    const int h   = blockIdx.x & 15;
    const float scale = 0.125f;   // 1/sqrt(64)

    const float* qbase = qkv + (size_t)b * 2048 * 3072 + h * 64;
    const float* kbase = qbase + 1024;
    const float* mrow  = mask + (size_t)b * 2048;

    for (int phase = 0; phase < 2; ++phase) {
        const float* base  = phase ? kbase : qbase;
        const float* wvecg = phase ? wb : wa;
        const float  bias  = phase ? bb[0] : ba[0];

        if (tid < 64) wv[tid] = wvecg[tid];
        __syncthreads();

        // scores + local max
        float lmax = -1e30f;
        for (int s = tid; s < 2048; s += 256) {
            const float4* rp4 = (const float4*)(base + (size_t)s * 3072);
            float dot = 0.0f;
            #pragma unroll
            for (int d4 = 0; d4 < 16; ++d4) {
                float4 xv = rp4[d4];
                if (phase) {
                    xv.x *= gv[4 * d4 + 0]; xv.y *= gv[4 * d4 + 1];
                    xv.z *= gv[4 * d4 + 2]; xv.w *= gv[4 * d4 + 3];
                }
                dot += xv.x * wv[4 * d4 + 0] + xv.y * wv[4 * d4 + 1] +
                       xv.z * wv[4 * d4 + 2] + xv.w * wv[4 * d4 + 3];
            }
            float scv = (dot + bias) * scale + mrow[s];
            sc[s] = scv;
            lmax = fmaxf(lmax, scv);
        }
        red[tid] = lmax; __syncthreads();
        for (int off = 128; off > 0; off >>= 1) {
            if (tid < off) red[tid] = fmaxf(red[tid], red[tid + off]);
            __syncthreads();
        }
        const float smax = red[0]; __syncthreads();

        // exp + sum
        float lsum = 0.0f;
        for (int s = tid; s < 2048; s += 256) {
            float e = __expf(sc[s] - smax);
            sc[s] = e;
            lsum += e;
        }
        red[tid] = lsum; __syncthreads();
        for (int off = 128; off > 0; off >>= 1) {
            if (tid < off) red[tid] += red[tid + off];
            __syncthreads();
        }
        const float inv = 1.0f / red[0]; __syncthreads();

        // weighted pooling: thread owns dim d, strides rows by 4
        const int d = tid & 63, grp = tid >> 6;
        float acc = 0.0f;
        for (int s = grp; s < 2048; s += 4) {
            float xv = base[(size_t)s * 3072 + d];
            if (phase) xv *= gv[d];
            acc += sc[s] * xv;
        }
        red[grp * 64 + d] = acc; __syncthreads();
        if (tid < 64) {
            float t = (red[tid] + red[64 + tid] + red[128 + tid] + red[192 + tid]) * inv;
            if (phase == 0) gv[tid] = t;
            else pav[(size_t)blockIdx.x * 64 + tid] = t;
        }
        __syncthreads();
    }
}

// u = v * p_av, emit f16 (layout [B,S,H*HD])
__global__ void gate_u(const float* __restrict__ qkv, const float* __restrict__ pav,
                       half_t* __restrict__ uh, int total)
{
    int i = blockIdx.x * blockDim.x + threadIdx.x;
    if (i >= total) return;
    const int row = i >> 10;       // b*S + s
    const int c   = i & 1023;      // h*64 + d
    const int b   = row >> 11;
    float v  = qkv[(size_t)row * 3072 + 2048 + c];
    float pv = pav[(size_t)b * 1024 + c];
    uh[i] = (half_t)(v * pv);
}

// LayerNorm over D=1024 per row
__launch_bounds__(256)
__global__ void layernorm_row(const float* __restrict__ hbuf,
                              const float* __restrict__ gamma,
                              const float* __restrict__ beta,
                              float* __restrict__ out)
{
    __shared__ float red[256];
    const int row = blockIdx.x, tid = threadIdx.x;
    const float4 v = ((const float4*)(hbuf + (size_t)row * 1024))[tid];

    red[tid] = v.x + v.y + v.z + v.w; __syncthreads();
    for (int off = 128; off > 0; off >>= 1) {
        if (tid < off) red[tid] += red[tid + off];
        __syncthreads();
    }
    const float mu = red[0] * (1.0f / 1024.0f); __syncthreads();

    float dx = v.x - mu, dy = v.y - mu, dz = v.z - mu, dw = v.w - mu;
    red[tid] = dx * dx + dy * dy + dz * dz + dw * dw; __syncthreads();
    for (int off = 128; off > 0; off >>= 1) {
        if (tid < off) red[tid] += red[tid + off];
        __syncthreads();
    }
    const float rs = rsqrtf(red[0] * (1.0f / 1024.0f) + 1e-6f);

    const float4 g = ((const float4*)gamma)[tid];
    const float4 bt = ((const float4*)beta)[tid];
    float4 o;
    o.x = dx * rs * g.x + bt.x;
    o.y = dy * rs * g.y + bt.y;
    o.z = dz * rs * g.z + bt.z;
    o.w = dw * rs * g.w + bt.w;
    ((float4*)(out + (size_t)row * 1024))[tid] = o;
}

// f32 -> f16 with optional column packing: dst[r*dstStride + dstOff + c] = src[r*cols + c]
__global__ void convert_pack(const float* __restrict__ src, half_t* __restrict__ dst,
                             int cols, int dstStride, int dstOff, long long total)
{
    long long i = (long long)blockIdx.x * blockDim.x + threadIdx.x;
    if (i >= total) return;
    int r = (int)(i / cols);
    int c = (int)(i % cols);
    dst[(size_t)r * dstStride + dstOff + c] = (half_t)src[i];
}

// ---------------------------------------------------------------------------
extern "C" void kernel_launch(void* const* d_in, const int* in_sizes, int n_in,
                              void* d_out, int out_size, void* d_ws, size_t ws_size,
                              hipStream_t stream)
{
    const float* x       = (const float*)d_in[0];
    const float* mask    = (const float*)d_in[1];
    const float* Wq      = (const float*)d_in[2];
    const float* bq      = (const float*)d_in[3];
    const float* Wk      = (const float*)d_in[4];
    const float* bk      = (const float*)d_in[5];
    const float* Wv      = (const float*)d_in[6];
    const float* bv      = (const float*)d_in[7];
    const float* wa      = (const float*)d_in[8];
    const float* ba      = (const float*)d_in[9];
    const float* wb      = (const float*)d_in[10];
    const float* bb      = (const float*)d_in[11];
    const float* Wu      = (const float*)d_in[12];
    const float* bu      = (const float*)d_in[13];
    const float* Wo      = (const float*)d_in[14];
    const float* bo      = (const float*)d_in[15];
    const float* gamma   = (const float*)d_in[16];
    const float* beta_ln = (const float*)d_in[17];
    float* out = (float*)d_out;

    char* w = (char*)d_ws;
    const size_t MB = 1ull << 20;
    half_t* xh   = (half_t*)(w + 0);         //  32 MB
    half_t* WqkvH= (half_t*)(w + 32  * MB);  //   6 MB
    half_t* WoH  = (half_t*)(w + 38  * MB);  //   2 MB
    half_t* WuH  = (half_t*)(w + 40  * MB);  //   8 KB
    float*  qkv  = (float*) (w + 41  * MB);  // 192 MB
    float*  pav  = (float*) (w + 233 * MB);  //  32 KB
    half_t* uh   = (half_t*)(w + 234 * MB);  //  32 MB
    half_t* newr = (half_t*)(w + 266 * MB);  //  32 MB
    float*  hbuf = (float*) (w + 298 * MB);  //  64 MB

    const int M = 8 * 2048;          // 16384 rows
    const long long nX = (long long)M * 1024;
    const long long nW = 1024 * 1024;

    // 1) precision conversion + QKV weight packing
    convert_pack<<<(unsigned)((nX + 255) / 256), 256, 0, stream>>>(x,  xh,    1024, 1024, 0,    nX);
    convert_pack<<<(unsigned)((nW + 255) / 256), 256, 0, stream>>>(Wq, WqkvH, 1024, 3072, 0,    nW);
    convert_pack<<<(unsigned)((nW + 255) / 256), 256, 0, stream>>>(Wk, WqkvH, 1024, 3072, 1024, nW);
    convert_pack<<<(unsigned)((nW + 255) / 256), 256, 0, stream>>>(Wv, WqkvH, 1024, 3072, 2048, nW);
    convert_pack<<<(unsigned)((nW + 255) / 256), 256, 0, stream>>>(Wo, WoH,   1024, 1024, 0,    nW);
    convert_pack<<<(64 * 64 + 255) / 256, 256, 0, stream>>>(Wu, WuH, 64, 64, 0, 64 * 64);

    // 2) fused QKV projection: [16384,1024] @ [1024,3072]
    dim3 g1(M / TM, 3072 / TN);
    wmma_gemm_f16<<<g1, 256, 0, stream>>>(xh, 1024, WqkvH, 3072, qkv, 3072, 1024,
                                          bq, bk, bv, nullptr, 0);

    // 3) additive-attention pooling (alpha then beta) -> p_av
    attn_pool<<<8 * 16, 256, 0, stream>>>(qkv, mask, wa, ba, wb, bb, pav);

    // 4) gate values: u = v * p_av
    gate_u<<<(M * 1024) / 256, 256, 0, stream>>>(qkv, pav, uh, M * 1024);

    // 5) per-head Wu GEMM, fused +bu +q, emit f16
    dim3 g2(M / 128, 16);
    wmma_gemm_wu<<<g2, 256, 0, stream>>>(uh, WuH, bu, qkv, newr);

    // 6) Wo projection, fused +bo +x residual
    dim3 g3(M / TM, 1024 / TN);
    wmma_gemm_f16<<<g3, 256, 0, stream>>>(newr, 1024, WoH, 1024, hbuf, 1024, 1024,
                                          bo, nullptr, nullptr, x, 1);

    // 7) LayerNorm
    layernorm_row<<<M, 256, 0, stream>>>(hbuf, gamma, beta_ln, out);
}
